// FlowLMMainWrapper_24902220382265
// MI455X (gfx1250) — matrix operand order repaired and checked by reading, not compiled, not run on hardware
//
#include <hip/hip_runtime.h>
#include <hip/hip_fp16.h>

typedef __attribute__((ext_vector_type(16))) _Float16 v16h;
typedef __attribute__((ext_vector_type(4)))  _Float16 v4h;
typedef __attribute__((ext_vector_type(2)))  _Float16 v2h;
typedef __attribute__((ext_vector_type(8)))  float    v8f;

#define NLAYERS 6
#define DMODEL  1024
#define NHEAD   16
#define DHEAD   64
#define MAXSEQ  4096
#define DFF     4096
#define LATENT  32
#define LTOT    544   // 512 text + 32 latent
#define TLEN    512

// ---------------------------------------------------------------- utilities
__device__ __forceinline__ float block_reduce_sum256(float v, float* red) {
    int tid = threadIdx.x;
    red[tid] = v; __syncthreads();
    for (int o = 128; o > 0; o >>= 1) {
        if (tid < o) red[tid] += red[tid + o];
        __syncthreads();
    }
    float r = red[0]; __syncthreads();
    return r;
}

__device__ __forceinline__ float gelu_exact(float x) {
    return 0.5f * x * (1.0f + erff(x * 0.70710678118654752f));
}

// ---------------------------------------------------------------- prep
__global__ void copy_f32v4_kernel(const float4* __restrict__ in, float4* __restrict__ out, int n4) {
    for (int i = blockIdx.x * blockDim.x + threadIdx.x; i < n4; i += gridDim.x * blockDim.x)
        out[i] = in[i];
}

// convert input fp16 caches into the float output cache regions (wide streaming)
__global__ void cache_copy_kernel(const __half2* __restrict__ in, float4* __restrict__ out, long n4) {
    long stride = (long)gridDim.x * blockDim.x;
    for (long i = blockIdx.x * (long)blockDim.x + threadIdx.x; i < n4; i += stride) {
        float2 a = __half22float2(in[2 * i]);
        float2 b = __half22float2(in[2 * i + 1]);
        out[i] = make_float4(a.x, a.y, b.x, b.y);
    }
}

// rows 512..543 of x : (nan->bos fixed seq) @ W_in + b_in
__global__ __launch_bounds__(256)
void prep_seq_kernel(const float* __restrict__ seq, const float* __restrict__ bos,
                     const float* __restrict__ W_in, const float* __restrict__ b_in,
                     float* __restrict__ x) {
    __shared__ float s[LATENT];
    int row = blockIdx.x;              // 0..31
    int tid = threadIdx.x;
    if (tid < LATENT) {
        float v = seq[row * LATENT + tid];
        s[tid] = (v != v) ? bos[tid] : v;
    }
    __syncthreads();
    for (int c = tid; c < DMODEL; c += 256) {
        float acc = b_in[c];
        #pragma unroll
        for (int k = 0; k < LATENT; ++k) acc += s[k] * W_in[k * DMODEL + c];
        x[(TLEN + row) * DMODEL + c] = acc;
    }
}

// ---------------------------------------------------------------- layernorm
__global__ __launch_bounds__(256)
void layernorm_kernel(const float* __restrict__ x, const float* __restrict__ w,
                      const float* __restrict__ b, float* __restrict__ out) {
    __shared__ float red[256];
    int row = blockIdx.x, tid = threadIdx.x;
    const float* xr = x + (long)row * DMODEL;
    float v[4]; float s = 0.f;
    #pragma unroll
    for (int i = 0; i < 4; ++i) { v[i] = xr[tid + i * 256]; s += v[i]; }
    float mu = block_reduce_sum256(s, red) * (1.0f / DMODEL);
    float vs = 0.f;
    #pragma unroll
    for (int i = 0; i < 4; ++i) { float d = v[i] - mu; vs += d * d; }
    float var = block_reduce_sum256(vs, red) * (1.0f / DMODEL);
    float inv = rsqrtf(var + 1e-5f);
    #pragma unroll
    for (int i = 0; i < 4; ++i) {
        int c = tid + i * 256;
        out[(long)row * DMODEL + c] = (v[i] - mu) * inv * w[c] + b[c];
    }
}

// ---------------------------------------------------------------- WMMA GEMM
// C[M,N] = act(A[M,K] @ B[K,N] + bias) + residual  (f32 in/out, f16 WMMA compute)
// 64x128 tile per block, 8 waves; wave w owns rows 16*(w&3).., cols 64*(w>>2)..
// Register-double-buffered pipeline: next K-tile global loads are issued after
// the barrier and consumed at the top of the next iteration, hiding global
// latency behind the WMMA + LDS work. N,K multiples of 128/32 (all call sites);
// M tail handled by clamping load rows (padding rows never stored).
__global__ __launch_bounds__(256)
void gemm_wmma_kernel(const float* __restrict__ A, int lda,
                      const float* __restrict__ B, int ldb,
                      const float* __restrict__ bias,
                      const float* __restrict__ residual, int ldr,
                      float* __restrict__ C, int ldc,
                      int M, int N, int K, int act) {
    __shared__ _Float16 As[64 * 40];    // row-major, padded stride 40
    __shared__ _Float16 Bs[128 * 40];   // column n major: Bs[n*40 + k]
    const int tid   = threadIdx.x;
    const int lane  = tid & 31;
    const int wave  = tid >> 5;
    const int tileN = blockIdx.x * 128;
    const int tileM = blockIdx.y * 64;
    const int wrow  = (wave & 3) << 4;
    const int wcol  = (wave >> 2) << 6;

    // ---- per-thread staging coordinates (fixed for whole kernel)
    int aRow[2], aCol[2];
    const float* aPtr[2];
    #pragma unroll
    for (int i = 0; i < 2; ++i) {
        int idx = i * 256 + tid;            // 0..511
        aRow[i] = idx >> 3;                 // 0..63
        aCol[i] = (idx & 7) << 2;           // 0..28
        int gm = tileM + aRow[i]; gm = gm < M ? gm : M - 1;   // clamp, no divergence
        aPtr[i] = A + (long)gm * lda + aCol[i];
    }
    int bK[2], bN[2];
    const float* bPtr[2];
    #pragma unroll
    for (int i = 0; i < 2; ++i) {
        int u = i * 256 + tid;              // 0..511 pair-units
        bK[i] = (u >> 5) << 1;              // even k: 0..30
        bN[i] = (u & 31) << 2;              // 0..124
        bPtr[i] = B + (long)bK[i] * ldb + tileN + bN[i];
    }

    // ---- prologue: load first K tile into registers
    float4 aReg[2], bReg0[2], bReg1[2];
    #pragma unroll
    for (int i = 0; i < 2; ++i) aReg[i] = *(const float4*)(aPtr[i]);
    #pragma unroll
    for (int i = 0; i < 2; ++i) {
        bReg0[i] = *(const float4*)(bPtr[i]);
        bReg1[i] = *(const float4*)(bPtr[i] + ldb);
    }

    v8f acc[4] = {v8f{}, v8f{}, v8f{}, v8f{}};

    for (int k0 = 0; k0 < K; k0 += 32) {
        // ---- dump staged registers to LDS (convert f32 -> f16)
        #pragma unroll
        for (int i = 0; i < 2; ++i) {
            v4h hv = { (_Float16)aReg[i].x, (_Float16)aReg[i].y,
                       (_Float16)aReg[i].z, (_Float16)aReg[i].w };
            *(v4h*)(&As[aRow[i] * 40 + aCol[i]]) = hv;
        }
        #pragma unroll
        for (int i = 0; i < 2; ++i) {
            v2h h0 = { (_Float16)bReg0[i].x, (_Float16)bReg1[i].x };
            v2h h1 = { (_Float16)bReg0[i].y, (_Float16)bReg1[i].y };
            v2h h2 = { (_Float16)bReg0[i].z, (_Float16)bReg1[i].z };
            v2h h3 = { (_Float16)bReg0[i].w, (_Float16)bReg1[i].w };
            _Float16* bp = &Bs[bN[i] * 40 + bK[i]];
            *(v2h*)(bp)       = h0;
            *(v2h*)(bp + 40)  = h1;
            *(v2h*)(bp + 80)  = h2;
            *(v2h*)(bp + 120) = h3;
        }
        __syncthreads();

        // ---- issue next tile's global loads (consumed next iteration)
        if (k0 + 32 < K) {
            #pragma unroll
            for (int i = 0; i < 2; ++i) {
                aPtr[i] += 32;
                aReg[i] = *(const float4*)(aPtr[i]);
            }
            #pragma unroll
            for (int i = 0; i < 2; ++i) {
                bPtr[i] += (long)32 * ldb;
                bReg0[i] = *(const float4*)(bPtr[i]);
                bReg1[i] = *(const float4*)(bPtr[i] + ldb);
            }
            if (k0 + 64 < K) {               // warm L2 one slab further out
                __builtin_prefetch(aPtr[0] + 32, 0, 3);
                __builtin_prefetch(bPtr[0] + (long)32 * ldb, 0, 3);
            }
        }

        // ---- A fragment (contiguous k = 16*(lane>>4) + j, matches B) -> 2x ds_load_b128
        v16h afrag;
        {
            int m  = wrow + (lane & 15);
            int kb = (lane >> 4) << 4;
            #pragma unroll
            for (int j = 0; j < 16; ++j) {
                int kk = kb + j;
                afrag[j] = As[m * 40 + kk];
            }
        }
        int kb16 = (lane >> 4) << 4;
        #pragma unroll
        for (int t = 0; t < 4; ++t) {
            v16h bfrag;
            int n = wcol + (t << 4) + (lane & 15);
            #pragma unroll
            for (int j = 0; j < 16; ++j) bfrag[j] = Bs[n * 40 + kb16 + j];
            acc[t] = __builtin_amdgcn_wmma_f32_16x16x32_f16(
                false, afrag, false, bfrag, (short)0, acc[t], false, false);
        }
        __syncthreads();
    }

    // ---- epilogue: VGPR r -> row wrow + (lane>=16 ? 8:0) + r ; col wcol + t*16 + (lane&15)
    #pragma unroll
    for (int t = 0; t < 4; ++t) {
        int gn = tileN + wcol + (t << 4) + (lane & 15);
        #pragma unroll
        for (int r = 0; r < 8; ++r) {
            int gm = tileM + wrow + ((lane >> 4) << 3) + r;
            if (gm < M) {
                float v = acc[t][r];
                if (bias)     v += bias[gn];
                if (act == 1) v  = gelu_exact(v);
                if (residual) v += residual[(long)gm * ldr + gn];
                C[(long)gm * ldc + gn] = v;
            }
        }
    }
}

// ---------------------------------------------------------------- RoPE + KV cache
// qkv row layout: [q(16x64) | k(16x64) | v(16x64)] = 3072 floats per token
__global__ __launch_bounds__(128)
void rope_cache_kernel(const float* __restrict__ qkv, const int* __restrict__ steps, int layer,
                       float* __restrict__ qout, float* __restrict__ kout, float* __restrict__ vout) {
    int l = blockIdx.x, h = blockIdx.y, tid = threadIdx.x;
    int pos = steps[layer] + l;
    if (pos >= MAXSEQ) return;
    const float* base = qkv + (long)l * (3 * DMODEL);
    const float* qp = base + h * DHEAD;
    const float* kp = base + DMODEL + h * DHEAD;
    const float* vp = base + 2 * DMODEL + h * DHEAD;
    long crow = ((long)pos * NHEAD + h) * DHEAD;
    if (tid < 32) {
        int j = tid;
        float freq = __powf(10000.0f, -(float)j / 32.0f);
        float ang  = (float)pos * freq;
        float c = __cosf(ang), s = __sinf(ang);
        float x1 = qp[2 * j], x2 = qp[2 * j + 1];
        long qrow = ((long)l * NHEAD + h) * DHEAD;
        qout[qrow + 2 * j]     = x1 * c - x2 * s;
        qout[qrow + 2 * j + 1] = x1 * s + x2 * c;
    } else if (tid < 64) {
        int j = tid - 32;
        float freq = __powf(10000.0f, -(float)j / 32.0f);
        float ang  = (float)pos * freq;
        float c = __cosf(ang), s = __sinf(ang);
        float x1 = kp[2 * j], x2 = kp[2 * j + 1];
        // fp16 round-trip like the reference cache
        kout[crow + 2 * j]     = __half2float(__float2half(x1 * c - x2 * s));
        kout[crow + 2 * j + 1] = __half2float(__float2half(x1 * s + x2 * c));
    } else {
        int d = tid - 64;
        vout[crow + d] = __half2float(__float2half(vp[d]));
    }
}

// ---------------------------------------------------------------- attention
__global__ __launch_bounds__(128)
void attention_kernel(const float* __restrict__ qbuf, const float* __restrict__ kc,
                      const float* __restrict__ vc, const int* __restrict__ steps, int layer,
                      float* __restrict__ out) {
    __shared__ float sc[4096];
    __shared__ float qs[DHEAD];
    __shared__ float red[128];
    int l = blockIdx.x, h = blockIdx.y, tid = threadIdx.x;
    int S = steps[layer] + l + 1;           // causal: kv_pos <= q_pos
    if (S > MAXSEQ) S = MAXSEQ;
    if (tid < DHEAD) qs[tid] = qbuf[((long)l * NHEAD + h) * DHEAD + tid];
    __syncthreads();

    const float scale = 0.125f;             // 1/sqrt(64)
    float mx = -3.4e38f;
    for (int s = tid; s < S; s += 128) {
        const float4* kr = (const float4*)(kc + ((long)s * NHEAD + h) * DHEAD);
        float d = 0.f;
        #pragma unroll
        for (int j = 0; j < DHEAD / 4; ++j) {
            float4 kv4 = kr[j];
            d += qs[4 * j] * kv4.x + qs[4 * j + 1] * kv4.y +
                 qs[4 * j + 2] * kv4.z + qs[4 * j + 3] * kv4.w;
        }
        d *= scale;
        sc[s] = d;
        mx = fmaxf(mx, d);
    }
    red[tid] = mx; __syncthreads();
    for (int o = 64; o > 0; o >>= 1) { if (tid < o) red[tid] = fmaxf(red[tid], red[tid + o]); __syncthreads(); }
    mx = red[0]; __syncthreads();

    float sum = 0.f;
    for (int s = tid; s < S; s += 128) { float e = __expf(sc[s] - mx); sc[s] = e; sum += e; }
    red[tid] = sum; __syncthreads();
    for (int o = 64; o > 0; o >>= 1) { if (tid < o) red[tid] += red[tid + o]; __syncthreads(); }
    float inv = 1.0f / red[0]; __syncthreads();

    if (tid < DHEAD) {
        float acc = 0.f;
        for (int s = 0; s < S; ++s) acc += sc[s] * vc[((long)s * NHEAD + h) * DHEAD + tid];
        out[((long)l * NHEAD + h) * DHEAD + tid] = acc * inv;
    }
}

// ---------------------------------------------------------------- final LN + eos + steps
__global__ __launch_bounds__(256)
void final_kernel(const float* __restrict__ x, const float* __restrict__ w,
                  const float* __restrict__ b, const float* __restrict__ W_eos,
                  const float* __restrict__ b_eos, const int* __restrict__ steps,
                  float* __restrict__ out, float* __restrict__ stepsOut) {
    __shared__ float red[256];
    int tid = threadIdx.x;
    const float* xr = x + (long)(LTOT - 1) * DMODEL;
    float v[4]; float s = 0.f;
    #pragma unroll
    for (int i = 0; i < 4; ++i) { v[i] = xr[tid + i * 256]; s += v[i]; }
    float mu = block_reduce_sum256(s, red) * (1.0f / DMODEL);
    float vs = 0.f;
    #pragma unroll
    for (int i = 0; i < 4; ++i) { float d = v[i] - mu; vs += d * d; }
    float var = block_reduce_sum256(vs, red) * (1.0f / DMODEL);
    float inv = rsqrtf(var + 1e-5f);
    float dot = 0.f;
    #pragma unroll
    for (int i = 0; i < 4; ++i) {
        int c = tid + i * 256;
        float y = (v[i] - mu) * inv * w[c] + b[c];
        out[c] = y;                          // conditioning
        dot += y * W_eos[c];
    }
    float e = block_reduce_sum256(dot, red);
    if (tid == 0) out[DMODEL] = e + b_eos[0];          // eos logit
    if (tid < NLAYERS) stepsOut[tid] = (float)(steps[tid] + LTOT);
}

// ---------------------------------------------------------------- launch
extern "C" void kernel_launch(void* const* d_in, const int* in_sizes, int n_in,
                              void* d_out, int out_size, void* d_ws, size_t ws_size,
                              hipStream_t stream) {
    const float*  seq   = (const float*) d_in[0];
    const float*  text  = (const float*) d_in[1];
    const __half* kcin  = (const __half*)d_in[2];
    const __half* vcin  = (const __half*)d_in[3];
    const int*    steps = (const int*)   d_in[4];
    const float*  bos   = (const float*) d_in[5];
    const float*  W_in  = (const float*) d_in[6];
    const float*  b_in  = (const float*) d_in[7];
    const float*  ln1w  = (const float*) d_in[8];
    const float*  ln1b  = (const float*) d_in[9];
    const float*  Wqkv  = (const float*) d_in[10];
    const float*  Wo    = (const float*) d_in[11];
    const float*  ln2w  = (const float*) d_in[12];
    const float*  ln2b  = (const float*) d_in[13];
    const float*  W1    = (const float*) d_in[14];
    const float*  b1    = (const float*) d_in[15];
    const float*  W2    = (const float*) d_in[16];
    const float*  b2    = (const float*) d_in[17];
    const float*  outnw = (const float*) d_in[18];
    const float*  outnb = (const float*) d_in[19];
    const float*  W_eos = (const float*) d_in[20];
    const float*  b_eos = (const float*) d_in[21];

    float* out = (float*)d_out;
    const long CACHE_N = (long)NLAYERS * MAXSEQ * NHEAD * DHEAD;   // 25,165,824
    float* kOut     = out + DMODEL + 1;
    float* vOut     = kOut + CACHE_N;
    float* stepsOut = vOut + CACHE_N;

    float* ws  = (float*)d_ws;
    float* x   = ws;                      // 544*1024
    float* h   = x   + LTOT * DMODEL;     // 544*1024
    float* qb  = h   + LTOT * DMODEL;     // 544*1024 (roped q)
    float* ao  = qb  + LTOT * DMODEL;     // 544*1024 (attention out)
    float* big = ao  + LTOT * DMODEL;     // 544*4096, shared by qkv & ff

    // prep: x = [text ; seq@W_in + b_in], cache passthrough copies
    copy_f32v4_kernel<<<256, 256, 0, stream>>>((const float4*)text, (float4*)x, TLEN * DMODEL / 4);
    prep_seq_kernel<<<32, 256, 0, stream>>>(seq, bos, W_in, b_in, x);
    cache_copy_kernel<<<4096, 256, 0, stream>>>((const __half2*)kcin, (float4*)kOut, CACHE_N / 4);
    cache_copy_kernel<<<4096, 256, 0, stream>>>((const __half2*)vcin, (float4*)vOut, CACHE_N / 4);

    const int MT = (LTOT + 63) / 64;                // 9 row tiles
    for (int i = 0; i < NLAYERS; ++i) {
        const long lc = (long)i * MAXSEQ * NHEAD * DHEAD;
        float* kL = kOut + lc;
        float* vL = vOut + lc;

        layernorm_kernel<<<LTOT, 256, 0, stream>>>(x, ln1w + i * DMODEL, ln1b + i * DMODEL, h);
        gemm_wmma_kernel<<<dim3(3 * DMODEL / 128, MT), 256, 0, stream>>>(
            h, DMODEL, Wqkv + (long)i * DMODEL * 3 * DMODEL, 3 * DMODEL,
            nullptr, nullptr, 0, big, 3 * DMODEL, LTOT, 3 * DMODEL, DMODEL, 0);
        rope_cache_kernel<<<dim3(LTOT, NHEAD), 128, 0, stream>>>(big, steps, i, qb, kL, vL);
        attention_kernel<<<dim3(LTOT, NHEAD), 128, 0, stream>>>(qb, kL, vL, steps, i, ao);
        gemm_wmma_kernel<<<dim3(DMODEL / 128, MT), 256, 0, stream>>>(
            ao, DMODEL, Wo + (long)i * DMODEL * DMODEL, DMODEL,
            nullptr, x, DMODEL, x, DMODEL, LTOT, DMODEL, DMODEL, 0);
        layernorm_kernel<<<LTOT, 256, 0, stream>>>(x, ln2w + i * DMODEL, ln2b + i * DMODEL, h);
        gemm_wmma_kernel<<<dim3(DFF / 128, MT), 256, 0, stream>>>(
            h, DMODEL, W1 + (long)i * DMODEL * DFF, DFF,
            b1 + (long)i * DFF, nullptr, 0, big, DFF, LTOT, DFF, DMODEL, 1 /*gelu*/);
        gemm_wmma_kernel<<<dim3(DMODEL / 128, MT), 256, 0, stream>>>(
            big, DFF, W2 + (long)i * DFF * DMODEL, DMODEL,
            b2 + (long)i * DMODEL, x, DMODEL, x, DMODEL, LTOT, DMODEL, DFF, 0);
    }

    final_kernel<<<1, 256, 0, stream>>>(x, outnw, outnb, W_eos, b_eos, steps, out, stepsOut);
}
